// NeuralCDE_70050916598424
// MI455X (gfx1250) — compile-verified, matrix operand order checked by reading, not run on hardware
//
#include <hip/hip_runtime.h>
#include <hip/hip_bf16.h>
#include <math.h>

typedef __attribute__((ext_vector_type(16))) _Float16 v16h;
typedef __attribute__((ext_vector_type(8)))  float    v8f;

#define B_    64
#define T_    64
#define IN_   32
#define H_    128
#define WID_  128
#define NSTEP 126          // (T-1)*SUBSTEPS

// ---- RK tableau (Tsit5-like, from reference) ----
__device__ __constant__ float ATAB[6][5] = {
  {0.f, 0.f, 0.f, 0.f, 0.f},
  {0.161f, 0.f, 0.f, 0.f, 0.f},
  {-0.008480655492356989f, 0.335480655492357f, 0.f, 0.f, 0.f},
  {2.8971530571054935f, -6.359448489975075f, 4.3622954328695815f, 0.f, 0.f},
  {5.325864828439257f, -11.748883564062828f, 7.4955393428898365f, -0.09249506636175525f, 0.f},
  {5.86145544294642f, -12.92096931784711f, 8.159367898576159f, -0.071584973281401f, -0.028269050394068383f}
};
__device__ __constant__ float CTAB[6] = {0.f, 0.161f, 0.327f, 0.9f, 0.9800255409045097f, 1.f};
__device__ __constant__ float BTAB[6] = {0.09646076681806523f, 0.01f, 0.4798896504144996f,
                                         1.379008574103742f, -3.290069515436081f, 2.324710524099774f};

// K-index mapping for 16-bit WMMA operand fragments (ISA 7.12.2):
// element j of lane l holds K = (j&7) + (j>=8 ? 16 : 0) + (l>=16 ? 8 : 0)
__device__ __forceinline__ int kmap(int lane, int j) {
  return (j & 7) + ((j & 8) << 1) + ((lane & 16) >> 1);
}

// Inverse mapping: flat f16 index of element (m, k) in a packed A/B-fragment
// buffer laid out as [ks][lane][j] (16 rows x 128 cols matrix).
__device__ __forceinline__ int apack_idx(int m, int k) {
  int ks = k >> 5, kk = k & 31;
  int lane = m + ((kk & 8) << 1);        // + 16 if kk in [8,16) or [24,32)
  int j = (kk & 7) | ((kk & 16) >> 1);   // + 8 if kk >= 16
  return (ks * 32 + lane) * 16 + j;
}

// Raw hardware transcendentals (v_exp_f32 / v_log_f32 are base-2; v_rcp_f32).
// Skips libm denorm guards — inputs here are tame (post-GEMM activations).
__device__ __forceinline__ float exp_fast(float x) {
  return __builtin_amdgcn_exp2f(x * 1.442695040888963f);
}
__device__ __forceinline__ float log_fast(float x) {
  return __builtin_amdgcn_logf(x) * 0.6931471805599453f;
}
__device__ __forceinline__ float tanh_fast(float x) {
  return 1.f - 2.f * __builtin_amdgcn_rcpf(exp_fast(2.f * x) + 1.f);
}
__device__ __forceinline__ float softplus_fast(float x) {
  float sp = log_fast(1.f + exp_fast(x));
  return (x > 20.f) ? x : sp;
}
__device__ __forceinline__ float sigmoid_fast(float x) {
  return __builtin_amdgcn_rcpf(1.f + exp_fast(-x));
}

// ------------------------------------------------------------------
// Prep: convert W2 (4096x128 f32 [n][k]) into fp16 packed B-operand
// fragments: tile t = nt*4+ks; layout [t][lane][j], so each lane reads
// its whole v16h with one coalesced 32B load.
// ------------------------------------------------------------------
__global__ void __launch_bounds__(256) pack_w2_kernel(const float* __restrict__ W2,
                                                      _Float16* __restrict__ w2p) {
  int g = blockIdx.x * blockDim.x + threadIdx.x;   // 0 .. 32767 (1024 tiles * 32 lanes)
  int tile = g >> 5, lane = g & 31;
  int nt = tile >> 2, ks = tile & 3;
  const float* src = W2 + (size_t)(nt * 16 + (lane & 15)) * WID_ + ks * 32;
  _Float16* dst = w2p + (size_t)g * 16;
#pragma unroll
  for (int j = 0; j < 16; ++j) dst[j] = (_Float16)src[kmap(lane, j)];
}

// ------------------------------------------------------------------
// Persistent per-batch-tile Neural-CDE integrator. 4 blocks x 256 thr.
// Block handles batch rows [16*blockIdx.x, +16) fully independently.
// ------------------------------------------------------------------
__global__ void __launch_bounds__(256, 1) ncde_kernel(
    const float* __restrict__ ts,
    const float* __restrict__ cd, const float* __restrict__ cc,
    const float* __restrict__ cb, const float* __restrict__ ca,
    const float* __restrict__ A0, const float* __restrict__ a0,
    const float* __restrict__ A1, const float* __restrict__ a1,
    const float* __restrict__ A2, const float* __restrict__ a2,
    const float* __restrict__ W0g, const float* __restrict__ b0g,
    const float* __restrict__ W1g, const float* __restrict__ b1g,
    const float* __restrict__ b2g,
    const _Float16* __restrict__ w2p,
    const float* __restrict__ Wl, const float* __restrict__ bl,
    float* __restrict__ out) {

  extern __shared__ __align__(32) char smem_raw[];
  // f16 region (all 32B-aligned offsets)
  _Float16* w0p   = (_Float16*)smem_raw;          // 16384 f16 (32 KB) packed B-frags
  _Float16* w1p   = w0p + 16384;                  // 16384 f16 (32 KB)
  _Float16* ypack = w1p + 16384;                  // 2048 f16: stage input, A-frag layout
  _Float16* hpack = ypack + 2048;                 // 2048 f16: h1, A-frag layout
  _Float16* gpack = hpack + 2048;                 // 2048 f16: h2, A-frag layout
  // f32 region
  float*    b0v   = (float*)(gpack + 2048);       // 128
  float*    b1v   = b0v + 128;                    // 128
  float*    ycur  = b1v + 128;                    // 16x128
  float*    dxl   = ycur + 2048;                  // 16x32 (dX; x0 at init)
  float*    kbuf  = dxl + 512;                    // 6 x 16x128 (encoder temps pre-loop)

  const int tid   = threadIdx.x;
  const int lane  = tid & 31;
  const int wv    = tid >> 5;            // wave 0..7
  const int brow0 = blockIdx.x * 16;
  const int mb    = (lane & 16) ? 8 : 0; // D-tile row base for this half-wave

  // ---- pack W0/W1 into LDS in B-fragment order ----
  for (int idx = tid; idx < 16384; idx += 256) {
    int j = idx & 15, ln = (idx >> 4) & 31, tile = idx >> 9;
    int nt = tile >> 2, ks = tile & 3;
    int n = nt * 16 + (ln & 15), k = ks * 32 + kmap(ln, j);
    w0p[idx] = (_Float16)W0g[n * WID_ + k];
    w1p[idx] = (_Float16)W1g[n * WID_ + k];
  }
  if (tid < 128) { b0v[tid] = b0g[tid]; b1v[tid] = b1g[tid]; }

  // ---- initial condition: x0 = coeff_a[:,0], 3-layer relu encoder ----
  float* enc1 = kbuf;            // 16x128 scratch (kbuf unused before time loop)
  float* enc2 = kbuf + 2048;     // 16x128 scratch
  for (int idx = tid; idx < 512; idx += 256) {
    int m = idx >> 5, i = idx & 31;
    dxl[idx] = ca[(size_t)(brow0 + m) * (T_ - 1) * IN_ + i];
  }
  __syncthreads();
  for (int idx = tid; idx < 2048; idx += 256) {           // h1 = relu(x0 A0^T + a0)
    int m = idx >> 7, n = idx & 127;
    float s = a0[n];
    for (int k = 0; k < IN_; ++k) s = fmaf(dxl[m * 32 + k], A0[n * IN_ + k], s);
    enc1[idx] = fmaxf(s, 0.f);
  }
  __syncthreads();
  for (int idx = tid; idx < 2048; idx += 256) {           // h2 = relu(h1 A1^T + a1)
    int m = idx >> 7, n = idx & 127;
    float s = a1[n];
    for (int k = 0; k < WID_; ++k) s = fmaf(enc1[m * 128 + k], A1[n * WID_ + k], s);
    enc2[idx] = fmaxf(s, 0.f);
  }
  __syncthreads();
  for (int idx = tid; idx < 2048; idx += 256) {           // y0 = h2 A2^T + a2
    int m = idx >> 7, n = idx & 127;
    float s = a2[n];
    for (int k = 0; k < WID_; ++k) s = fmaf(enc2[m * 128 + k], A2[n * WID_ + k], s);
    ycur[idx] = s;
  }
  __syncthreads();

  const float t0 = ts[0];
  const float dt = (ts[T_ - 1] - ts[0]) / (float)NSTEP;
  const v16h* w0v = (const v16h*)w0p;
  const v16h* w1v = (const v16h*)w1p;
  const v16h* yv  = (const v16h*)ypack;
  const v16h* hv  = (const v16h*)hpack;
  const v16h* gv  = (const v16h*)gpack;
  const v16h* w2v = (const v16h*)w2p;

  for (int step = 0; step < NSTEP; ++step) {
    float t = t0 + dt * (float)step;

    for (int s = 0; s < 6; ++s) {
      // --- stage input: ypack = f16(y + dt * sum_j a_sj k_j), fragment layout ---
      for (int idx = tid; idx < 2048; idx += 256) {
        float acc = ycur[idx];
        for (int j = 0; j < s; ++j) acc = fmaf(dt * ATAB[s][j], kbuf[j * 2048 + idx], acc);
        ypack[apack_idx(idx >> 7, idx & 127)] = (_Float16)acc;
      }
      // --- spline derivative dX(t_s)  (ts is uniform arange) ---
      float tt = t + CTAB[s] * dt;
      int   it = min(max((int)floorf(tt - t0), 0), T_ - 2);
      float fr = (tt - t0) - (float)it;
      for (int idx = tid; idx < 512; idx += 256) {
        int m = idx >> 5, i = idx & 31;
        size_t o = ((size_t)(brow0 + m) * (T_ - 1) + it) * IN_ + i;
        dxl[idx] = cb[o] + fr * (2.f * cc[o] + 3.f * fr * cd[o]);
      }
      __syncthreads();

      // ---- GEMM1: hpack = softplus(ystg @ W0^T + b0); one 16x16 N-tile/wave ----
      {
        v8f acc = {};
#pragma unroll
        for (int ks = 0; ks < 4; ++ks) {
          v16h a = yv[ks * 32 + lane];
          v16h b = w0v[(wv * 4 + ks) * 32 + lane];
          acc = __builtin_amdgcn_wmma_f32_16x16x32_f16(false, a, false, b, (short)0, acc, false, false);
        }
        int n = wv * 16 + (lane & 15);
        float bias = b0v[n];
#pragma unroll
        for (int v = 0; v < 8; ++v)
          hpack[apack_idx(mb + v, n)] = (_Float16)softplus_fast(acc[v] + bias);
      }
      __syncthreads();

      // ---- GEMM2: gpack = softplus(h1 @ W1^T + b1) ----
      {
        v8f acc = {};
#pragma unroll
        for (int ks = 0; ks < 4; ++ks) {
          v16h a = hv[ks * 32 + lane];
          v16h b = w1v[(wv * 4 + ks) * 32 + lane];
          acc = __builtin_amdgcn_wmma_f32_16x16x32_f16(false, a, false, b, (short)0, acc, false, false);
        }
        int n = wv * 16 + (lane & 15);
        float bias = b1v[n];
#pragma unroll
        for (int v = 0; v < 8; ++v)
          gpack[apack_idx(mb + v, n)] = (_Float16)softplus_fast(acc[v] + bias);
      }
      __syncthreads();

      // ---- GEMM3 (16x4096x128) + tanh + einsum over IN=32 -> k_s ----
      {
        float* kout = kbuf + s * 2048;
        v16h afr[4];
#pragma unroll
        for (int ks = 0; ks < 4; ++ks) afr[ks] = gv[ks * 32 + lane];

        for (int pt = 0; pt < 16; ++pt) {   // tile pair = one h column group
          int nt0 = wv * 32 + pt * 2;
          if (pt < 15)
            __builtin_prefetch((const void*)&w2v[(size_t)(nt0 + 2) * 4 * 32 + lane], 0, 0);
          // bias loads issued first so they overlap the WMMA chain
          float bias0 = b2g[nt0 * 16 + (lane & 15)];
          float bias1 = b2g[(nt0 + 1) * 16 + (lane & 15)];
          // two independent accumulation chains, interleaved per k-step
          v8f acc0 = {}, acc1 = {};
#pragma unroll
          for (int ks = 0; ks < 4; ++ks) {
            v16h b0f = w2v[((size_t)nt0 * 4 + ks) * 32 + lane];
            v16h b1f = w2v[((size_t)(nt0 + 1) * 4 + ks) * 32 + lane];
            acc0 = __builtin_amdgcn_wmma_f32_16x16x32_f16(false, afr[ks], false, b0f, (short)0, acc0, false, false);
            acc1 = __builtin_amdgcn_wmma_f32_16x16x32_f16(false, afr[ks], false, b1f, (short)0, acc1, false, false);
          }
          int hcol = wv * 16 + pt;
#pragma unroll
          for (int v = 0; v < 8; ++v) {
            int m = mb + v;
            float p = tanh_fast(acc0[v] + bias0) * dxl[m * 32 + (lane & 15)]
                    + tanh_fast(acc1[v] + bias1) * dxl[m * 32 + 16 + (lane & 15)];
            p += __shfl_xor(p, 1, 16);
            p += __shfl_xor(p, 2, 16);
            p += __shfl_xor(p, 4, 16);
            p += __shfl_xor(p, 8, 16);
            if ((lane & 15) == 0) kout[m * 128 + hcol] = p;
          }
        }
      }
      __syncthreads();
    }

    // --- y update: y += dt * sum_j b_j k_j ---
    for (int idx = tid; idx < 2048; idx += 256) {
      float acc = ycur[idx];
#pragma unroll
      for (int j = 0; j < 6; ++j) acc = fmaf(dt * BTAB[j], kbuf[j * 2048 + idx], acc);
      ycur[idx] = acc;
    }
    __syncthreads();
  }

  // ---- head: sigmoid(y @ Wl^T + bl), OUT=1 ----
  if (tid < 16) {
    float s = bl[0];
    for (int h = 0; h < H_; ++h) s = fmaf(ycur[tid * 128 + h], Wl[h], s);
    out[brow0 + tid] = sigmoid_fast(s);
  }
}

extern "C" void kernel_launch(void* const* d_in, const int* in_sizes, int n_in,
                              void* d_out, int out_size, void* d_ws, size_t ws_size,
                              hipStream_t stream) {
  const float* ts = (const float*)d_in[0];
  const float* cd = (const float*)d_in[1];
  const float* cc = (const float*)d_in[2];
  const float* cb = (const float*)d_in[3];
  const float* ca = (const float*)d_in[4];
  const float* A0 = (const float*)d_in[5];
  const float* a0 = (const float*)d_in[6];
  const float* A1 = (const float*)d_in[7];
  const float* a1 = (const float*)d_in[8];
  const float* A2 = (const float*)d_in[9];
  const float* a2 = (const float*)d_in[10];
  const float* W0 = (const float*)d_in[11];
  const float* b0 = (const float*)d_in[12];
  const float* W1 = (const float*)d_in[13];
  const float* b1 = (const float*)d_in[14];
  const float* W2 = (const float*)d_in[15];
  const float* b2 = (const float*)d_in[16];
  const float* Wl = (const float*)d_in[17];
  const float* bl = (const float*)d_in[18];

  _Float16* w2p = (_Float16*)d_ws;   // 4096*128 f16 = 1 MB packed W2

  pack_w2_kernel<<<128, 256, 0, stream>>>(W2, w2p);

  // dynamic LDS: 64KB packed weights + 3x4KB f16 activations + f32 state/k's
  size_t shbytes = (size_t)(16384 * 2 + 2048 * 3) * sizeof(_Float16)
                 + (size_t)(128 + 128 + 2048 + 512 + 6 * 2048) * sizeof(float);
  ncde_kernel<<<4, 256, shbytes, stream>>>(ts, cd, cc, cb, ca, A0, a0, A1, a1, A2, a2,
                                           W0, b0, W1, b1, b2, w2p, Wl, bl, (float*)d_out);
}
